// MultiHeadAttention_28776280883716
// MI455X (gfx1250) — matrix operand order repaired and checked by reading, not compiled, not run on hardware
//
#include <hip/hip_runtime.h>
#include <hip/hip_bf16.h>

// MI455X (gfx1250) flash-attention forward with f16 WMMA (f32 accumulate).
// B=8, S=1024, D=1024, H=16, dph=64, scale=sqrt(64)=8, p_drop=0.1
// Dropout tile (dominant HBM stream, 536 MB) staged via async global->LDS DMA.

typedef __attribute__((ext_vector_type(16))) _Float16 v16h;
typedef __attribute__((ext_vector_type(8)))  _Float16 v8h;
typedef __attribute__((ext_vector_type(8)))  float    v8f;
typedef __attribute__((ext_vector_type(4)))  float    v4f;
typedef int vi4 __attribute__((vector_size(16)));   // matches builtin's pointee type

#define AS1 __attribute__((address_space(1)))
#define AS3 __attribute__((address_space(3)))

#define WMMA_F16(A, B, C) \
    __builtin_amdgcn_wmma_f32_16x16x32_f16(false, (A), false, (B), (short)0, (C), false, false)

namespace {
constexpr int Bn  = 8;
constexpr int Sn  = 1024;
constexpr int Dn  = 1024;
constexpr int Hn  = 16;
constexpr int DPH = 64;     // head dim
constexpr int LDH = 72;     // f16 LDS row stride in halves (144B, 16B-aligned)
constexpr int LDD = 68;     // f32 dropout-tile row stride in floats (272B, 16B-aligned)
constexpr float NEG_INF_BIAS = -1e20f;
}

// Async copy of 16 bytes global -> LDS (ASYNCcnt-tracked DMA, no VGPR round-trip).
__device__ __forceinline__ void async_copy_b128(const float* gptr, float* lptr) {
#if __has_builtin(__builtin_amdgcn_global_load_async_to_lds_b128)
    __builtin_amdgcn_global_load_async_to_lds_b128(
        (AS1 vi4*)gptr, (AS3 vi4*)lptr, 0, 0);
#else
    asm volatile("global_load_async_to_lds_b128 %0, %1, off"
                 :: "v"((unsigned)(unsigned long long)(AS3 void*)lptr),
                    "v"((unsigned long long)gptr)
                 : "memory");
#endif
}

__device__ __forceinline__ void wait_asynccnt0() {
#if __has_builtin(__builtin_amdgcn_s_wait_asynccnt)
    __builtin_amdgcn_s_wait_asynccnt(0);
#else
    asm volatile("s_wait_asynccnt 0" ::: "memory");
#endif
}

// A-matrix 16x32 f16 fragment: lane(0-15) row=lane, halves K = hi*8+0..7 then 16+hi*8+0..7.
// p must point at [row*LDH + chunkBase + hi*8]
__device__ __forceinline__ v16h ld_a_frag(const _Float16* p) {
    v8h lo = *reinterpret_cast<const v8h*>(p);
    v8h hv = *reinterpret_cast<const v8h*>(p + 16);
    return __builtin_shufflevector(lo, hv, 0,1,2,3,4,5,6,7,8,9,10,11,12,13,14,15);
}

// B-matrix 32x16 f16 fragment: lane(0-15) col N=lane, halves K = hi*8+0..7 then 16+hi*8+0..7.
// p must point at [col*LDH + chunkBase + hi*8]
__device__ __forceinline__ v16h ld_b_frag(const _Float16* p) {
    v8h lo = *reinterpret_cast<const v8h*>(p);
    v8h hv = *reinterpret_cast<const v8h*>(p + 16);
    return __builtin_shufflevector(lo, hv, 0,1,2,3,4,5,6,7,8,9,10,11,12,13,14,15);
}

__global__ void __launch_bounds__(128)
mha_flash_wmma_kernel(const float* __restrict__ Q, const float* __restrict__ K,
                      const float* __restrict__ V, const int* __restrict__ Msk,
                      const float* __restrict__ Drop, float* __restrict__ Out)
{
    __shared__ __align__(16) _Float16 sQ [64 * LDH];
    __shared__ __align__(16) _Float16 sK [64 * LDH];
    __shared__ __align__(16) _Float16 sVt[64 * LDH];   // transposed: [dph][key]
    __shared__ __align__(16) _Float16 sP [4][16 * LDH];
    __shared__ __align__(16) float    sD [64 * LDD];   // f32 dropout tile [qrow][key]
    __shared__ float sBias[64];

    const int qt = blockIdx.x & 15;        // S/64 = 16 q-tiles
    const int bh = blockIdx.x >> 4;        // 0..127
    const int b  = bh >> 4;                // H = 16
    const int h  = bh & 15;
    const int q0 = qt * 64;

    const int tid  = threadIdx.x;
    const int wave = tid >> 5;
    const int lane = tid & 31;
    const int l16  = lane & 15;
    const int hi   = lane >> 4;

    // ---------------- stage Q tile (scaled by 1/8, f32 -> f16) ----------------
    {
        const float* qbase = Q + ((size_t)(b * Sn + q0)) * Dn + h * DPH;
        #pragma unroll
        for (int i = 0; i < 8; ++i) {
            int linear = tid + 128 * i;                // 1024 float4s
            int row = linear >> 4;
            int c4  = (linear & 15) * 4;
            v4f x = *reinterpret_cast<const v4f*>(qbase + (size_t)row * Dn + c4);
            _Float16* dst = &sQ[row * LDH + c4];
            #pragma unroll
            for (int e = 0; e < 4; ++e) dst[e] = (_Float16)(x[e] * 0.125f);
        }
    }
    __syncthreads();

    // per-wave Q A-fragments (live in registers for the whole kernel)
    const _Float16* qp = &sQ[(wave * 16 + l16) * LDH + hi * 8];
    const v16h qa0 = ld_a_frag(qp);
    const v16h qa1 = ld_a_frag(qp + 32);

    // online-softmax state (row m = r + 8*hi of this wave's slab)
    float mrow[8], lrow[8];
    v8f oacc[4];
    #pragma unroll
    for (int r = 0; r < 8; ++r) { mrow[r] = -1e30f; lrow[r] = 0.0f; }
    #pragma unroll
    for (int no = 0; no < 4; ++no)
        #pragma unroll
        for (int r = 0; r < 8; ++r) oacc[no][r] = 0.0f;

    const size_t kvbase0 = ((size_t)b * Sn) * Dn + h * DPH;
    const float inv_keep = 1.0f / 0.9f;
    const float* dtile0 = Drop + ((size_t)bh * Sn + q0) * Sn;   // [qrow][key] view

    for (int kt = 0; kt < Sn / 64; ++kt) {
        __syncthreads();   // previous iteration's LDS reads complete

        // ------------- async DMA: dropout tile (64x64 f32) -> LDS -------------
        const float* dtile = dtile0 + kt * 64;
        #pragma unroll
        for (int i = 0; i < 8; ++i) {
            int linear = tid + 128 * i;
            int row = linear >> 4;
            int c4  = (linear & 15) * 4;
            async_copy_b128(dtile + (size_t)row * Sn + c4, &sD[row * LDD + c4]);
        }

        // ------------- stage K (row-major) and V (transposed) as f16 -------------
        const float* kbase = K + kvbase0 + (size_t)kt * 64 * Dn;
        const float* vbase = V + kvbase0 + (size_t)kt * 64 * Dn;
        #pragma unroll
        for (int i = 0; i < 8; ++i) {
            int linear = tid + 128 * i;
            int row = linear >> 4;
            int c4  = (linear & 15) * 4;
            v4f xk = *reinterpret_cast<const v4f*>(kbase + (size_t)row * Dn + c4);
            _Float16* dk = &sK[row * LDH + c4];
            #pragma unroll
            for (int e = 0; e < 4; ++e) dk[e] = (_Float16)xk[e];
            v4f xv = *reinterpret_cast<const v4f*>(vbase + (size_t)row * Dn + c4);
            #pragma unroll
            for (int e = 0; e < 4; ++e) sVt[(c4 + e) * LDH + row] = (_Float16)xv[e];
        }
        if (tid < 64)
            sBias[tid] = (Msk[b * Sn + kt * 64 + tid] == 0) ? NEG_INF_BIAS : 0.0f;

        wait_asynccnt0();      // this wave's DMA landed in LDS
        __syncthreads();       // everyone's staging visible

        // ------------- S = Q K^T (16 x 64 per wave), + mask bias -------------
        v8f sc[4];
        #pragma unroll
        for (int n = 0; n < 4; ++n) {
            const _Float16* kp = &sK[(n * 16 + l16) * LDH + hi * 8];
            v16h kb0 = ld_b_frag(kp);
            v16h kb1 = ld_b_frag(kp + 32);
            v8f acc;
            #pragma unroll
            for (int r = 0; r < 8; ++r) acc[r] = 0.0f;
            acc = WMMA_F16(qa0, kb0, acc);
            acc = WMMA_F16(qa1, kb1, acc);
            float bias = sBias[n * 16 + l16];
            #pragma unroll
            for (int r = 0; r < 8; ++r) acc[r] += bias;
            sc[n] = acc;
        }

        // ------------- online softmax (row stats via 16-lane shfl_xor) -------------
        float mnew[8], alpha[8], lsum[8];
        #pragma unroll
        for (int r = 0; r < 8; ++r) {
            float v0 = fmaxf(fmaxf(sc[0][r], sc[1][r]), fmaxf(sc[2][r], sc[3][r]));
            #pragma unroll
            for (int off = 1; off < 16; off <<= 1)
                v0 = fmaxf(v0, __shfl_xor(v0, off, 32));
            mnew[r]  = fmaxf(mrow[r], v0);
            alpha[r] = __expf(mrow[r] - mnew[r]);
            lsum[r]  = 0.0f;
        }
        #pragma unroll
        for (int n = 0; n < 4; ++n)
            #pragma unroll
            for (int r = 0; r < 8; ++r) {
                float p = __expf(sc[n][r] - mnew[r]);
                sc[n][r] = p;
                lsum[r] += p;
            }
        #pragma unroll
        for (int r = 0; r < 8; ++r) {
            float s = lsum[r];
            #pragma unroll
            for (int off = 1; off < 16; off <<= 1)
                s += __shfl_xor(s, off, 32);
            lrow[r] = lrow[r] * alpha[r] + s;   // denominator WITHOUT dropout
            mrow[r] = mnew[r];
        }
        #pragma unroll
        for (int no = 0; no < 4; ++no)
            #pragma unroll
            for (int r = 0; r < 8; ++r) oacc[no][r] *= alpha[r];

        // ------------- apply dropout from LDS, spill P (f16) to per-wave LDS -------------
        // C-layout element (n, r, lane) -> tile row = wave*16 + r + 8*hi, col = n*16 + l16.
        // LDS read: half-waves hit rows 8 apart -> bank offset 32, conflict-free.
        _Float16* sPw = sP[wave];
        #pragma unroll
        for (int n = 0; n < 4; ++n)
            #pragma unroll
            for (int r = 0; r < 8; ++r) {
                int row = r + 8 * hi;
                int col = n * 16 + l16;
                float dm = sD[(wave * 16 + row) * LDD + col];
                sPw[row * LDH + col] = (_Float16)(sc[n][r] * dm * inv_keep);
            }
        asm volatile("s_wait_dscnt 0" ::: "memory");  // wave-local store->load on sP

        // ------------- O += P V -------------
        const _Float16* pA = &sPw[l16 * LDH + hi * 8];
        v16h pa0 = ld_a_frag(pA);
        v16h pa1 = ld_a_frag(pA + 32);
        #pragma unroll
        for (int no = 0; no < 4; ++no) {
            const _Float16* vp = &sVt[(no * 16 + l16) * LDH + hi * 8];
            v16h vb0 = ld_b_frag(vp);
            v16h vb1 = ld_b_frag(vp + 32);
            oacc[no] = WMMA_F16(pa0, vb0, oacc[no]);
            oacc[no] = WMMA_F16(pa1, vb1, oacc[no]);
        }
    }

    // ------------- epilogue: O / l -> out[b, q, h*64 + d] -------------
    float* obase = Out + ((size_t)(b * Sn + q0 + wave * 16)) * Dn + h * DPH;
    #pragma unroll
    for (int r = 0; r < 8; ++r) {
        int row = r + 8 * hi;
        float inv_l = 1.0f / lrow[r];
        #pragma unroll
        for (int no = 0; no < 4; ++no)
            obase[(size_t)row * Dn + no * 16 + l16] = oacc[no][r] * inv_l;
    }
}

extern "C" void kernel_launch(void* const* d_in, const int* in_sizes, int n_in,
                              void* d_out, int out_size, void* d_ws, size_t ws_size,
                              hipStream_t stream) {
    const float* q    = (const float*)d_in[0];
    const float* k    = (const float*)d_in[1];
    const float* v    = (const float*)d_in[2];
    const int*   msk  = (const int*)d_in[3];
    const float* drop = (const float*)d_in[4];
    float*       out  = (float*)d_out;

    dim3 grid(Bn * Hn * (Sn / 64));   // 2048 blocks: (b, h, q-tile)
    dim3 block(128);                  // 4 wave32s, one 16-row Q slab each
    mha_flash_wmma_kernel<<<grid, block, 0, stream>>>(q, k, v, msk, drop, out);
}
